// Faster_RCNN_81922206204092
// MI455X (gfx1250) — matrix operand order repaired and checked by reading, not compiled, not run on hardware
//
#include <hip/hip_runtime.h>
#include <hip/hip_bf16.h>
#include <math.h>

typedef __bf16 bf16;
typedef bf16  v16bf __attribute__((ext_vector_type(16)));
typedef float v8f   __attribute__((ext_vector_type(8)));
typedef unsigned int u32x4 __attribute__((ext_vector_type(4)));
typedef int          i32x4 __attribute__((ext_vector_type(4)));
typedef int          i32x8 __attribute__((ext_vector_type(8)));

#define C_CH   512
#define H_F    37
#define W_F    50
#define PP     7
#define N_ROIS 300
#define M_PAD  320        // 300 rounded up to 20 tiles of 16
#define K1     25088      // 512*7*7, divisible by 32
#define DH     4096
#define NCLS   21

#define TILE_K   32
#define TILE_N   128
#define LDS_ROW  129      // 128 dwords + 1 dword TDM pad -> bank-conflict-free halves
#define TILE_DW  (TILE_K * LDS_ROW)   // dwords per buffered tile

// ---------------------------------------------------------------------------
// Kernel 1: ROI adaptive max pool (PyTorch bin edges) -> bf16 A1[M_PAD][K1]
// ---------------------------------------------------------------------------
__global__ void __launch_bounds__(256)
roi_pool_kernel(const float* __restrict__ feat,   // [512,37,50]
                const float* __restrict__ rois,   // [300,4] (y1,x1,y2,x2)
                bf16* __restrict__ A1) {
  const int roi = blockIdx.x;
  bf16* out = A1 + (size_t)roi * K1;
  if (roi >= N_ROIS) {                 // zero-pad rows 300..319
    for (int i = threadIdx.x; i < K1; i += blockDim.x) out[i] = (bf16)0.0f;
    return;
  }
  const float inv = 1.0f / 16.0f;
  const int y1 = (int)(rois[roi * 4 + 0] * inv);
  const int x1 = (int)(rois[roi * 4 + 1] * inv);
  const int y2 = (int)(rois[roi * 4 + 2] * inv);
  const int x2 = (int)(rois[roi * 4 + 3] * inv);
  const int hs = y2 - y1 + 1, wsp = x2 - x1 + 1;
  for (int i = threadIdx.x; i < K1; i += blockDim.x) {
    const int c  = i / 49;
    const int b  = i - c * 49;
    const int ph = b / 7, pw = b - ph * 7;
    int ys = y1 + (ph * hs) / PP;
    int ye = y1 + ((ph + 1) * hs + PP - 1) / PP;
    int xs = x1 + (pw * wsp) / PP;
    int xe = x1 + ((pw + 1) * wsp + PP - 1) / PP;
    if (ye > H_F) ye = H_F;
    if (xe > W_F) xe = W_F;
    float m = -3.402823466e38f;
    const float* fc = feat + (size_t)c * (H_F * W_F);
    for (int y = ys; y < ye; ++y)
      for (int x = xs; x < xe; ++x) {
        const float v = fc[y * W_F + x];
        m = v > m ? v : m;
      }
    out[i] = (bf16)m;
  }
}

// K index inside a 16x32 bf16 A fragment for vector element e (half = lane>>4)
__device__ __forceinline__ int a_kidx(int e) {
  return ((e >> 3) << 4) + (((e >> 1) & 3) << 1) + (e & 1);
}

// ---------------------------------------------------------------------------
// TDM: DMA one [TILE_K x TILE_N] f32 weight tile into LDS (padded rows).
// D# packing per CDNA5 ISA §8.3/8.4; groups 2/3 zero (2-D tensor).
// ---------------------------------------------------------------------------
__device__ __forceinline__ void tdm_load_tile(const float* gsrc, unsigned lds_byte,
                                              int Nn, int K) {
  const unsigned long long ga = (unsigned long long)(uintptr_t)gsrc;
  u32x4 g0;
  g0[0] = 1u;                                      // count=1, user descriptor
  g0[1] = lds_byte;                                // lds_addr (bytes)
  g0[2] = (unsigned)(ga & 0xFFFFFFFFu);            // global_addr[31:0]
  g0[3] = (unsigned)((ga >> 32) & 0x01FFFFFFu)     // global_addr[56:32]
          | 0x80000000u;                           // type=2 ("image")
  i32x8 g1;
  // data_size=4B (2), pad_enable=1, pad_interval=6 (128 dw), pad_amount=0 (1 dw)
  g1[0] = (2 << 16) | (1 << 20) | (6 << 22);
  g1[1] = (int)(((unsigned)Nn & 0xFFFFu) << 16);                     // tensor_dim0 lo16
  g1[2] = (int)(((unsigned)Nn >> 16) | (((unsigned)K & 0xFFFFu) << 16)); // dim0 hi16 | dim1 lo16
  g1[3] = (int)(((unsigned)K >> 16) | ((unsigned)TILE_N << 16));     // dim1 hi16 | tile_dim0
  g1[4] = TILE_K;                                                    // tile_dim1 (tile_dim2=0)
  g1[5] = Nn;                                                        // tensor_dim0_stride lo32
  g1[6] = 0;                                                         // stride hi | dim1_stride lo
  g1[7] = 0;
  const i32x4 gz = {0, 0, 0, 0};
#if __clang_major__ >= 23
  const i32x8 gz8 = {0, 0, 0, 0, 0, 0, 0, 0};
  __builtin_amdgcn_tensor_load_to_lds(g0, g1, gz, gz, gz8, 0);
#else
  __builtin_amdgcn_tensor_load_to_lds(g0, g1, gz, gz, 0);
#endif
}

// ---------------------------------------------------------------------------
// Kernel 2/3: Out = leaky_relu(A(bf16) x W(f32, TDM-streamed) + bias), bf16
// Block: 256 thr (8 waves). Tile 32M x 128N. Wave w owns cols w*16..w*16+15
// with two 16x16 accumulators; each B fragment feeds 2 WMMAs.
// Weights double-buffered in LDS via tensor_load_to_lds + s_wait_tensorcnt.
// ---------------------------------------------------------------------------
__global__ void __launch_bounds__(256)
gemm_bf16_kernel(const bf16*  __restrict__ A,    // [M_PAD][K] bf16
                 const float* __restrict__ Wt,   // [K][Nn] f32 row-major
                 const float* __restrict__ bias, // [Nn]
                 bf16*        __restrict__ Out,  // [M_PAD][Nn] bf16
                 int K, int Nn, float slope) {
  __shared__ float smem[2 * TILE_DW];   // two TDM tiles (~33 KB)
  const int tid   = threadIdx.x;
  const int lane  = tid & 31;
  const int wave  = tid >> 5;
  const int half  = lane >> 4;
  const int lrow  = lane & 15;
  const int nbase = blockIdx.x * TILE_N;
  const int mbase = blockIdx.y * 32;

  v8f c0 = {}; v8f c1 = {};
  const bf16* a0p = A + (size_t)(mbase + lrow) * K;
  const bf16* a1p = A + (size_t)(mbase + 16 + lrow) * K;

  // Prologue: stream first weight tile.
  if (wave == 0) {
    tdm_load_tile(&Wt[(size_t)0 * Nn + nbase], 0u, Nn, K);
    __builtin_amdgcn_s_wait_tensorcnt(0);
  }
  __syncthreads();

  int buf = 0;
  for (int kb = 0; kb < K; kb += TILE_K, buf ^= 1) {
    // Kick off DMA of the next tile into the other buffer (overlaps compute).
    if (wave == 0 && kb + TILE_K < K)
      tdm_load_tile(&Wt[(size_t)(kb + TILE_K) * Nn + nbase],
                    (unsigned)((buf ^ 1) * TILE_DW * 4), Nn, K);

    // A fragments (per-lane global bf16 loads; ISA 16-bit A layout -> b128s)
    v16bf a0, a1;
    #pragma unroll
    for (int e = 0; e < 16; ++e) {
      const int kk = kb + a_kidx(e) + half * 8;
      a0[e] = a0p[kk];
      a1[e] = a1p[kk];
    }
    // B fragment: read f32 tile from LDS, convert to bf16 on the fly.
    const float* tile = &smem[buf * TILE_DW];
    const int nl = wave * 16 + lrow;
    v16bf bfrag;
    #pragma unroll
    for (int e = 0; e < 16; ++e)
      bfrag[e] = (bf16)tile[(a_kidx(e) + 8 * half) * LDS_ROW + nl];

    c0 = __builtin_amdgcn_wmma_f32_16x16x32_bf16(false, a0, false, bfrag,
                                                 (short)0, c0, false, false);
    c1 = __builtin_amdgcn_wmma_f32_16x16x32_bf16(false, a1, false, bfrag,
                                                 (short)0, c1, false, false);

    // Drain this iteration's TDM before the whole block flips buffers.
    if (wave == 0) __builtin_amdgcn_s_wait_tensorcnt(0);
    __syncthreads();
  }

  // Epilogue: C/D layout -> lanes 0-15: M=r, lanes 16-31: M=r+8; N = lane%16
  const int ncol = nbase + wave * 16 + lrow;
  const float bv = bias[ncol];
  #pragma unroll
  for (int r = 0; r < 8; ++r) {
    const int row0 = mbase + r + half * 8;
    float v = c0[r] + bv;
    v = v > 0.0f ? v : v * slope;
    Out[(size_t)row0 * Nn + ncol] = (bf16)v;
    const int row1 = mbase + 16 + r + half * 8;
    float w = c1[r] + bv;
    w = w > 0.0f ? w : w * slope;
    Out[(size_t)row1 * Nn + ncol] = (bf16)w;
  }
}

// ---------------------------------------------------------------------------
// Kernel 4: per-ROI heads: 25 dot products (4 box + 21 class), softmax(21)
// ---------------------------------------------------------------------------
__global__ void __launch_bounds__(256)
heads_kernel(const bf16*  __restrict__ A3,   // [M_PAD][4096] bf16
             const float* __restrict__ Wb,   // [4096][4]
             const float* __restrict__ bb,   // [4]
             const float* __restrict__ Wc,   // [4096][21]
             const float* __restrict__ bc,   // [21]
             float* __restrict__ out_boxes,  // [300][4]
             float* __restrict__ out_cls) {  // [300][21]
  __shared__ float red[256];
  __shared__ float res[25];
  const int roi = blockIdx.x;
  const bf16* h = A3 + (size_t)roi * DH;

  float acc[25];
  #pragma unroll
  for (int j = 0; j < 25; ++j) acc[j] = 0.0f;

  for (int k = threadIdx.x; k < DH; k += 256) {
    const float hv = (float)h[k];
    #pragma unroll
    for (int j = 0; j < 4; ++j)  acc[j]     += hv * Wb[(size_t)k * 4 + j];
    #pragma unroll
    for (int j = 0; j < 21; ++j) acc[4 + j] += hv * Wc[(size_t)k * 21 + j];
  }
  for (int o = 0; o < 25; ++o) {
    red[threadIdx.x] = acc[o];
    __syncthreads();
    for (int s = 128; s > 0; s >>= 1) {
      if (threadIdx.x < s) red[threadIdx.x] += red[threadIdx.x + s];
      __syncthreads();
    }
    if (threadIdx.x == 0) res[o] = red[0];
    __syncthreads();
  }
  if (threadIdx.x == 0) {
    for (int j = 0; j < 4; ++j) out_boxes[roi * 4 + j] = res[j] + bb[j];
    float logits[NCLS], mx = -3.402823466e38f;
    for (int j = 0; j < NCLS; ++j) {
      logits[j] = res[4 + j] + bc[j];
      mx = fmaxf(mx, logits[j]);
    }
    float s = 0.0f;
    for (int j = 0; j < NCLS; ++j) { logits[j] = __expf(logits[j] - mx); s += logits[j]; }
    const float invs = 1.0f / s;
    for (int j = 0; j < NCLS; ++j) out_cls[roi * NCLS + j] = logits[j] * invs;
  }
}

// ---------------------------------------------------------------------------
extern "C" void kernel_launch(void* const* d_in, const int* in_sizes, int n_in,
                              void* d_out, int out_size, void* d_ws, size_t ws_size,
                              hipStream_t stream) {
  const float* feat = (const float*)d_in[0];  // img_features [1,512,37,50]
  const float* rois = (const float*)d_in[1];  // proposal_rois [300,4]
  const float* W1   = (const float*)d_in[2];  // [25088,4096]
  const float* b1   = (const float*)d_in[3];
  const float* W2   = (const float*)d_in[4];  // [4096,4096]
  const float* b2   = (const float*)d_in[5];
  const float* Wb   = (const float*)d_in[6];  // [4096,4]
  const float* bb   = (const float*)d_in[7];
  const float* Wc   = (const float*)d_in[8];  // [4096,21]
  const float* bc   = (const float*)d_in[9];

  char* ws = (char*)d_ws;
  bf16* A1 = (bf16*)ws;                                            // 320*25088*2 B
  bf16* A2 = (bf16*)(ws + (size_t)M_PAD * K1 * 2);                 // 320*4096*2 B
  bf16* A3 = (bf16*)(ws + (size_t)M_PAD * K1 * 2 + (size_t)M_PAD * DH * 2);

  float* out_boxes = (float*)d_out;            // [300,4]
  float* out_cls   = out_boxes + N_ROIS * 4;   // [300,21]

  roi_pool_kernel<<<M_PAD, 256, 0, stream>>>(feat, rois, A1);
  gemm_bf16_kernel<<<dim3(DH / TILE_N, M_PAD / 32), 256, 0, stream>>>(
      A1, W1, b1, A2, K1, DH, 0.1f);
  gemm_bf16_kernel<<<dim3(DH / TILE_N, M_PAD / 32), 256, 0, stream>>>(
      A2, W2, b2, A3, DH, DH, 0.1f);
  heads_kernel<<<N_ROIS, 256, 0, stream>>>(A3, Wb, bb, Wc, bc, out_boxes, out_cls);
}
